// RecurrentGCN_66503273611817
// MI455X (gfx1250) — matrix operand order repaired and checked by reading, not compiled, not run on hardware
//
#include <hip/hip_runtime.h>

// ---------------- problem constants (from reference) ----------------
#define N_CH 62
#define NB   128
#define NN   (N_CH * NB)   // 7936 nodes
#define FDIM 32
#define TDIM 128
#define FT   (FDIM * TDIM) // 4096
#define HID  512
#define G3   (3 * HID)     // 1536
#define DIN  (N_CH * FDIM) // 1984

typedef __attribute__((ext_vector_type(16))) __bf16 v16bf;
typedef __attribute__((ext_vector_type(8)))  float  v8f;
typedef __attribute__((address_space(3))) __bf16 lds_bf16;

// ---------------- WMMA helpers (CDNA5 bf16 16x16x32, f32 accum) ----------------
__device__ __forceinline__ v8f wmma_bf16(v16bf a, v16bf b, v8f c) {
  // (neg_a, A, neg_b, B, c_mod, C, reuse_a, reuse_b)
  return __builtin_amdgcn_wmma_f32_16x16x32_bf16(false, a, false, b, (short)0, c,
                                                 false, false);
}

// A-matrix 16x32 fragment: lane m=lane&15; element e -> K =
// e + (lane>=16 ? 16 : 0). With strideK==1 each lane reads one contiguous
// 16-element run.
template <typename T>
__device__ __forceinline__ v16bf load_a(const T* p, int strideM, int strideK, int lane) {
  int m = lane & 15;
  int koff = (lane & 16) ? 16 : 0;
  v16bf a;
#pragma unroll
  for (int e = 0; e < 16; ++e) {
    int k = e + koff;
    a[e] = (__bf16)(float)p[m * strideM + k * strideK];
  }
  return a;
}

// B-matrix 32x16 fragment: lane n=lane&15; element e -> K = e + (lane>=16 ? 16 : 0).
// With strideK==1 each lane reads one contiguous 16-element run.
template <typename T>
__device__ __forceinline__ v16bf load_b(const T* p, int strideK, int strideN, int lane) {
  int n = lane & 15;
  int koff = (lane & 16) ? 16 : 0;
  v16bf b;
#pragma unroll
  for (int e = 0; e < 16; ++e) {
    int k = e + koff;
    b[e] = (__bf16)(float)p[k * strideK + n * strideN];
  }
  return b;
}

// B fragment gathering node-rows of X[N][4096] for graph g, zero-padded past 62 rows
template <typename T>
__device__ __forceinline__ v16bf load_b_nodes(const T* X, int g, int k0, int col0, int lane) {
  int n = lane & 15;
  int koff = (lane & 16) ? 16 : 0;
  v16bf r;
#pragma unroll
  for (int e = 0; e < 16; ++e) {
    int k = k0 + e + koff;
    float v = (k < N_CH) ? (float)X[(long)(g * N_CH + k) * FT + col0 + n] : 0.0f;
    r[e] = (__bf16)v;
  }
  return r;
}

// ---------------- small utility kernels ----------------
__global__ void zero_f32_k(float* p, int n) {
  int i = blockIdx.x * blockDim.x + threadIdx.x;
  if (i < n) p[i] = 0.0f;
}
__global__ void zero_bf16_k(__bf16* p, int n) {
  int i = blockIdx.x * blockDim.x + threadIdx.x;
  if (i < n) p[i] = (__bf16)0.0f;
}
__global__ void cvt_bf16_k(const float* __restrict__ s, __bf16* __restrict__ d, int n) {
  int i = blockIdx.x * blockDim.x + threadIdx.x;
  if (i < n) d[i] = (__bf16)s[i];
}

// degree accumulation (segment_sum over src)
__global__ void deg_k(const int* __restrict__ ei, const float* __restrict__ w,
                      float* __restrict__ deg, int E) {
  int e = blockIdx.x * blockDim.x + threadIdx.x;
  if (e < E) atomicAdd(&deg[ei[e]], w[e]);
}
__global__ void dinv_k(float* p, int n) {
  int i = blockIdx.x * blockDim.x + threadIdx.x;
  if (i < n) p[i] = (p[i] > 0.0f) ? rsqrtf(p[i]) : 0.0f;
}
// scatter normalized edge weights into dense per-graph 64x64 Laplacian (L[dst][src])
__global__ void adj_k(const int* __restrict__ ei, const float* __restrict__ w,
                      const float* __restrict__ dinv, float* __restrict__ AhatF, int E) {
  int e = blockIdx.x * blockDim.x + threadIdx.x;
  if (e >= E) return;
  int s = ei[e], d = ei[E + e];
  float wn = -dinv[s] * w[e] * dinv[d];
  int g = s / N_CH;
  int sl = s - g * N_CH;
  int dl = d - g * N_CH;
  atomicAdd(&AhatF[((long)g * 64 + dl) * 64 + sl], wn);
}

// ---------------- Chebyshev propagation: Y = Ahat_g @ X_g (batched 64x4096x64) ----
// TX2MODE: Y = 2*(Ahat@X) - Z   (Tx2 recurrence)
template <typename TX, typename TZ, bool TX2MODE>
__global__ void cheb_prop_k(const __bf16* __restrict__ A, const TX* __restrict__ X,
                            const TZ* __restrict__ Z, __bf16* __restrict__ Y) {
  int nt = blockIdx.x;   // 0..255 column tile (f*t flattened)
  int mt = blockIdx.y;   // 0..3 node tile
  int g  = blockIdx.z;   // graph
  int lane = threadIdx.x;
  v8f c = {};
  const __bf16* Ab = A + (long)g * 4096 + mt * 16 * 64;
#pragma unroll
  for (int kc = 0; kc < 2; ++kc) {
    v16bf af = load_a(Ab + kc * 32, 64, 1, lane);
    v16bf bf = load_b_nodes(X, g, kc * 32, nt * 16, lane);
    c = wmma_bf16(af, bf, c);
  }
  int n = lane & 15;
  int mb = (lane & 16) ? 8 : 0;
  int col = nt * 16 + n;
#pragma unroll
  for (int i = 0; i < 8; ++i) {
    int m = mt * 16 + mb + i;
    if (m < N_CH) {
      long idx = (long)(g * N_CH + m) * FT + col;
      float y = c[i];
      if (TX2MODE) y = 2.0f * y - (float)Z[idx];
      Y[idx] = (__bf16)y;
    }
  }
}

// ---------------- Chebyshev feature mix: out[n,g,t] = sum_k sum_f Txk[n,f,t] Wk[f,g] + b
// TOSEQ: store transposed into GRU input layout seq[b][t][c*32+g] (bf16)
template <typename T0, bool LEAKY, bool TOSEQ>
__global__ void cheb_mix_k(const T0* __restrict__ X0, const __bf16* __restrict__ X1,
                           const __bf16* __restrict__ X2, const float* __restrict__ W,
                           const float* __restrict__ bias, __bf16* __restrict__ out) {
  int tt = blockIdx.x;        // 0..7 time tile
  int gt = blockIdx.y;        // 0..1 out-feature tile
  int node = blockIdx.z;      // 0..7935
  int lane = threadIdx.x;
  v8f c = {};
  // A(m=g, k=f) = W[kterm][f*32 + g]  -> strideM=1, strideK=32
  {
    v16bf af = load_a(W + 0 * 1024 + gt * 16, 1, 32, lane);
    v16bf bf = load_b(X0 + (long)node * FT + tt * 16, TDIM, 1, lane);
    c = wmma_bf16(af, bf, c);
  }
  {
    v16bf af = load_a(W + 1 * 1024 + gt * 16, 1, 32, lane);
    v16bf bf = load_b(X1 + (long)node * FT + tt * 16, TDIM, 1, lane);
    c = wmma_bf16(af, bf, c);
  }
  {
    v16bf af = load_a(W + 2 * 1024 + gt * 16, 1, 32, lane);
    v16bf bf = load_b(X2 + (long)node * FT + tt * 16, TDIM, 1, lane);
    c = wmma_bf16(af, bf, c);
  }
  int n = lane & 15;
  int mb = (lane & 16) ? 8 : 0;
  int t = tt * 16 + n;
#pragma unroll
  for (int i = 0; i < 8; ++i) {
    int gg = gt * 16 + mb + i;
    float v = c[i] + bias[gg];
    if (LEAKY) v = (v > 0.0f) ? v : 0.01f * v;
    if (TOSEQ) {
      int b = node / N_CH, ch = node - (node / N_CH) * N_CH;
      out[((long)b * TDIM + t) * DIN + ch * FDIM + gg] = (__bf16)v;
    } else {
      out[(long)node * FT + gg * TDIM + t] = (__bf16)v;
    }
  }
}

// ---------------- GRU input GEMM: xg[16384,1536] = seq @ W_ih^T + b_ih ----------
// LDS-staged tiled WMMA GEMM, 8 waves/block on a 32x64 output tile.
// A tile (seq rows) staged via GLOBAL_LOAD_ASYNC_TO_LDS_B128 (ASYNCcnt-tracked,
// no VGPR round-trip); B tile stored [n][k] so fragments are contiguous K runs.
__global__ void xg_gemm_k(const __bf16* __restrict__ seq, const __bf16* __restrict__ WihB,
                          const float* __restrict__ bih, float* __restrict__ xg) {
  __shared__ __bf16 As[32 * 32];  // [m][k]
  __shared__ __bf16 Bs[64 * 32];  // [n][k]  (transposed stage -> contiguous B frags)
  int m0 = blockIdx.x * 32;
  int n0 = blockIdx.y * 64;
  int tid = threadIdx.x;          // 256 threads = 8 waves
  int wid = tid >> 5, lane = tid & 31;
  int wm = wid >> 2, wn = wid & 3;
  // 32-bit LDS offset of As for the async tile loads
  unsigned as_base = (unsigned)(unsigned long long)(lds_bf16*)As;
  int ar  = tid >> 2;             // 0..63 (only 0..31 used)
  int seg = tid & 3;              // 16-byte segment within a 64-byte row
  v8f c = {};
  for (int kc = 0; kc < DIN / 32; ++kc) {   // 62 k-chunks
    int k0 = kc * 32;
    if (kc + 1 < DIN / 32)  // prefetch next A chunk (global_prefetch_b8)
      __builtin_prefetch(&seq[(long)(m0 + (tid & 31)) * DIN + k0 + 32], 0, 1);
    // --- A: 32 rows x 64B, async global->LDS, one B128 per lane of waves 0..3 ---
    if (tid < 128) {
      unsigned long long ga =
          (unsigned long long)(const void*)&seq[(long)(m0 + ar) * DIN + k0 + seg * 8];
      unsigned la = as_base + (unsigned)(ar * 64 + seg * 16);
      asm volatile("global_load_async_to_lds_b128 %0, %1, off"
                   :: "v"(la), "v"(ga) : "memory");
    }
    // --- B: W_ih^T tile staged [n][k]; 8 contiguous bf16 per thread ---
    {
      int nn = tid >> 2;
      int kk = (tid & 3) * 8;
#pragma unroll
      for (int e = 0; e < 8; ++e)
        Bs[nn * 32 + kk + e] = WihB[(long)(n0 + nn) * DIN + k0 + kk + e];
    }
    asm volatile("s_wait_asynccnt 0x0" ::: "memory");
    __syncthreads();
    v16bf af = load_a(As + wm * 16 * 32, 32, 1, lane);
    v16bf bf = load_b(Bs + wn * 16 * 32, 1, 32, lane);
    c = wmma_bf16(af, bf, c);
    __syncthreads();
  }
  int n = lane & 15;
  int mb = (lane & 16) ? 8 : 0;
  int gn = n0 + wn * 16 + n;
  float bv = bih[gn];
#pragma unroll
  for (int i = 0; i < 8; ++i) {
    int gm = m0 + wm * 16 + mb + i;
    xg[(long)gm * G3 + gn] = c[i] + bv;
  }
}

// ---------------- fused GRU step (one of 128 sequential steps) -------------------
// One wave computes the r/z/n 16x16 tiles at the same (batch-tile, j-tile):
// hg = h @ W_hh^T + b_hh (K=512, 48 WMMAs, shared A fragment), then gates in regs.
// W_hh pre-converted to bf16 [1536][512]: B fragments are contiguous 32B runs.
__global__ void gru_step_k(const __bf16* __restrict__ hin, __bf16* __restrict__ hout,
                           const float* __restrict__ xg, const __bf16* __restrict__ WhhB,
                           const float* __restrict__ bhh, int t) {
  int bt = blockIdx.x;  // 0..7  batch tile
  int jt = blockIdx.y;  // 0..31 hidden tile
  int lane = threadIdx.x;
  v8f cr = {}, cz = {}, cn = {};
#pragma unroll 4
  for (int kc = 0; kc < HID / 32; ++kc) {   // 16 chunks
    int k0 = kc * 32;
    v16bf af = load_a(hin + (long)(bt * 16) * HID + k0, HID, 1, lane);
    // B(k,n) = WhhB[(gate*512 + jt*16 + n)*512 + k] -> strideK=1 (contiguous)
    v16bf br = load_b(WhhB + (long)(0 * HID + jt * 16) * HID + k0, 1, HID, lane);
    v16bf bz = load_b(WhhB + (long)(1 * HID + jt * 16) * HID + k0, 1, HID, lane);
    v16bf bn = load_b(WhhB + (long)(2 * HID + jt * 16) * HID + k0, 1, HID, lane);
    cr = wmma_bf16(af, br, cr);
    cz = wmma_bf16(af, bz, cz);
    cn = wmma_bf16(af, bn, cn);
  }
  int n = lane & 15;
  int mb = (lane & 16) ? 8 : 0;
  int col = jt * 16 + n;
  float bhr = bhh[col], bhz = bhh[HID + col], bhn = bhh[2 * HID + col];
#pragma unroll
  for (int i = 0; i < 8; ++i) {
    int b = bt * 16 + mb + i;
    long xrow = ((long)b * TDIM + t) * G3;
    float xr = xg[xrow + col];
    float xz = xg[xrow + HID + col];
    float xn = xg[xrow + 2 * HID + col];
    float r = 1.0f / (1.0f + __expf(-(xr + cr[i] + bhr)));
    float z = 1.0f / (1.0f + __expf(-(xz + cz[i] + bhz)));
    float nn = tanhf(xn + r * (cn[i] + bhn));
    float hold = (float)hin[(long)b * HID + col];
    hout[(long)b * HID + col] = (__bf16)((1.0f - z) * nn + z * hold);
  }
}

// ---------------- final linear: out[128,4] = h @ W_lin^T + b_lin -----------------
__global__ void final_lin_k(const __bf16* __restrict__ h, const float* __restrict__ Wl,
                            const float* __restrict__ bl, float* __restrict__ out) {
  int tid = blockIdx.x * blockDim.x + threadIdx.x;
  if (tid >= NB * 4) return;
  int b = tid >> 2, o = tid & 3;
  float s = bl[o];
#pragma unroll 8
  for (int k = 0; k < HID; ++k) s += (float)h[(long)b * HID + k] * Wl[o * HID + k];
  out[tid] = s;
}

// ---------------- host launcher ----------------
extern "C" void kernel_launch(void* const* d_in, const int* in_sizes, int n_in,
                              void* d_out, int out_size, void* d_ws, size_t ws_size,
                              hipStream_t stream) {
  const float* x    = (const float*)d_in[0];
  const int*   ei   = (const int*)d_in[1];
  const float* ew   = (const float*)d_in[2];
  const float* Wc1  = (const float*)d_in[4];
  const float* bc1  = (const float*)d_in[5];
  const float* Wc2  = (const float*)d_in[6];
  const float* bc2  = (const float*)d_in[7];
  const float* Wih  = (const float*)d_in[8];
  const float* Whh  = (const float*)d_in[9];
  const float* bih  = (const float*)d_in[10];
  const float* bhh  = (const float*)d_in[11];
  const float* Wlin = (const float*)d_in[12];
  const float* blin = (const float*)d_in[13];
  float* out = (float*)d_out;
  const int E = in_sizes[2];  // 126976

  // workspace layout (256B aligned); xg overlays the dead T1+T2 region
  char* ws = (char*)d_ws;
  size_t off = 0;
  auto carve = [&](size_t bytes) {
    void* p = ws + off;
    off = (off + bytes + 255) & ~(size_t)255;
    return p;
  };
  float*  deg   = (float*)carve((size_t)NN * 4);
  float*  AhatF = (float*)carve((size_t)NB * 64 * 64 * 4);
  __bf16* AhatB = (__bf16*)carve((size_t)NB * 64 * 64 * 2);
  __bf16* WihB  = (__bf16*)carve((size_t)G3 * DIN * 2);   // 6 MB
  __bf16* WhhB  = (__bf16*)carve((size_t)G3 * HID * 2);   // 1.5 MB
  __bf16* hping = (__bf16*)carve((size_t)NB * HID * 2);
  __bf16* hpong = (__bf16*)carve((size_t)NB * HID * 2);
  __bf16* T1    = (__bf16*)carve((size_t)NN * FT * 2);    // 65 MB
  __bf16* T2    = (__bf16*)carve((size_t)NN * FT * 2);    // 65 MB
  __bf16* H1    = (__bf16*)carve((size_t)NN * FT * 2);    // 65 MB
  __bf16* SEQ   = (__bf16*)carve((size_t)NB * TDIM * DIN * 2); // 65 MB
  float*  XG    = (float*)T1;  // 100 MB overlay on T1+T2 (dead after conv2 mix)

  // ---- graph normalization: deg -> dinv -> dense per-graph Laplacian (bf16) ----
  zero_f32_k<<<(NN + 255) / 256, 256, 0, stream>>>(deg, NN);
  zero_f32_k<<<(NB * 4096 + 255) / 256, 256, 0, stream>>>(AhatF, NB * 4096);
  deg_k<<<(E + 255) / 256, 256, 0, stream>>>(ei, ew, deg, E);
  dinv_k<<<(NN + 255) / 256, 256, 0, stream>>>(deg, NN);
  adj_k<<<(E + 255) / 256, 256, 0, stream>>>(ei, ew, deg, AhatF, E);
  cvt_bf16_k<<<(NB * 4096 + 255) / 256, 256, 0, stream>>>(AhatF, AhatB, NB * 4096);
  // one-time bf16 conversion of the GRU weight matrices
  cvt_bf16_k<<<(G3 * DIN + 255) / 256, 256, 0, stream>>>(Wih, WihB, G3 * DIN);
  cvt_bf16_k<<<(G3 * HID + 255) / 256, 256, 0, stream>>>(Whh, WhhB, G3 * HID);

  dim3 gProp(FT / 16, 4, NB);   // 256 x 4 x 128, one wave per 16x16 tile
  dim3 gMix(TDIM / 16, 2, NN);  // 8 x 2 x 7936

  // ---- ChebConv 1 (leaky-relu epilogue) ----
  cheb_prop_k<float, float, false><<<gProp, 32, 0, stream>>>(AhatB, x, x, T1);
  cheb_prop_k<__bf16, float, true><<<gProp, 32, 0, stream>>>(AhatB, T1, x, T2);
  cheb_mix_k<float, true, false><<<gMix, 32, 0, stream>>>(x, T1, T2, Wc1, bc1, H1);

  // ---- ChebConv 2 (stores transposed bf16 GRU input) ----
  cheb_prop_k<__bf16, __bf16, false><<<gProp, 32, 0, stream>>>(AhatB, H1, H1, T1);
  cheb_prop_k<__bf16, __bf16, true><<<gProp, 32, 0, stream>>>(AhatB, T1, H1, T2);
  cheb_mix_k<__bf16, false, true><<<gMix, 32, 0, stream>>>(H1, T1, T2, Wc2, bc2, SEQ);

  // ---- GRU input GEMM (async-LDS staged WMMA) ----
  xg_gemm_k<<<dim3((NB * TDIM) / 32, G3 / 64), 256, 0, stream>>>(SEQ, WihB, bih, XG);

  // ---- GRU recurrence: 128 sequential fused steps, ping-pong hidden state ----
  zero_bf16_k<<<(NB * HID + 255) / 256, 256, 0, stream>>>(hping, NB * HID);
  __bf16* hc = hping;
  __bf16* hn = hpong;
  for (int t = 0; t < TDIM; ++t) {
    gru_step_k<<<dim3(NB / 16, HID / 16), 32, 0, stream>>>(hc, hn, XG, WhhB, bhh, t);
    __bf16* tmp = hc; hc = hn; hn = tmp;
  }

  // ---- output head ----
  final_lin_k<<<2, 256, 0, stream>>>(hc, Wlin, blin, out);

  (void)n_in; (void)out_size; (void)ws_size;
}
